// GCNSpatioTemporalAttention_2430951490188
// MI455X (gfx1250) — compile-verified, matrix operand order checked by reading, not run on hardware
//
#include <hip/hip_runtime.h>

typedef __attribute__((ext_vector_type(16))) _Float16 v16h;
typedef __attribute__((ext_vector_type(8)))  float    v8f;

#define NN  50000
#define TT  12
#define DD  32
#define HHD 8
#define EE  800000

static __device__ __forceinline__ v8f wmma32(v16h a, v16h b, v8f c) {
  // v_wmma_f32_16x16x32_f16 : D = A(16x32) x B(32x16) + C(16x16 f32)
  return __builtin_amdgcn_wmma_f32_16x16x32_f16(false, a, false, b, (short)0, c, false, false);
}

// A fragment from an LDS-resident f32 [16][32] tile (row-major).
// Lane L: row M = L%16; K = kb..kb+7 and kb+16..kb+23 with kb = (L/16)*8.
static __device__ __forceinline__ v16h afrag_lds(const float* buf, int lane) {
  int row = lane & 15, kb = (lane >> 4) * 8;
  const float* p = buf + row * 32 + kb;
  v16h a;
#pragma unroll
  for (int i = 0; i < 8; ++i) { a[i] = (_Float16)p[i]; a[i + 8] = (_Float16)p[i + 16]; }
  return a;
}

// Swizzle helper: element i (0..15) of lane l's B-fragment for column-tile tl
// of a [32 x ld] row-major weight matrix maps to W[row][col]:
//   kb = (l/16)*8, col = (l%16) + tl*16, row = kb + (i<8 ? i : i+8)
static __device__ __forceinline__ void swizzle_weights_to_frags(
    _Float16* dst, const float* W, int ld, int ntiles, int tid, int nthreads) {
  int total = ntiles * 512;  // per tile: 32 lanes x 16 halves
  for (int f = tid; f < total; f += nthreads) {
    int tl = f >> 9;
    int r2 = f & 511;
    int l  = r2 >> 4;
    int i  = r2 & 15;
    int kb  = (l >> 4) * 8;
    int col = (l & 15) + (tl << 4);
    int row = kb + ((i < 8) ? i : i + 8);
    dst[f] = (_Float16)W[row * ld + col];
  }
}

// ---------------------------------------------------------------------------
// Kernel 1: fused temporal stack. One wave = one node. x read once from HBM,
// all intermediates stay in LDS. Weights pre-swizzled to fragment layout so
// each B-fragment is 2x ds_load_b128.
// ---------------------------------------------------------------------------
__global__ __launch_bounds__(128) void k_temporal(
    const float* __restrict__ x,
    const float* __restrict__ Win, const float* __restrict__ bin,
    const float* __restrict__ Wq,  const float* __restrict__ bq,
    const float* __restrict__ Wk,  const float* __restrict__ bk,
    const float* __restrict__ Wv,  const float* __restrict__ bv,
    const float* __restrict__ Wo,  const float* __restrict__ bo,
    const float* __restrict__ lng, const float* __restrict__ lnb,
    float* __restrict__ xn) {
  __shared__ v16h  sWf[5 * 2 * 32];       // 5 matrices x 2 col-tiles x 32 lanes (10KB)
  __shared__ float sB[5 * 32];
  __shared__ float sLN[64];
  __shared__ float sbuf[4 * 5 * 512];     // per-wave: x1/o, x2, q/out, k, v (40KB)

  int tid = threadIdx.x, lane = tid & 31, wid = tid >> 5;
  const float* Wsrc[5] = {Win, Wq, Wk, Wv, Wo};
  const float* Bsrc[5] = {bin, bq, bk, bv, bo};
#pragma unroll
  for (int m = 0; m < 5; ++m)
    swizzle_weights_to_frags((_Float16*)(sWf + m * 64), Wsrc[m], 32, 2, tid, 128);
  for (int i = tid; i < 5 * 32; i += 128) sB[i] = Bsrc[i >> 5][i & 31];
  for (int i = tid; i < 64; i += 128)     sLN[i] = (i < 32) ? lng[i] : lnb[i - 32];
  __syncthreads();

  int n = blockIdx.x * 4 + wid;           // grid sized so n < NN always
  float* bx1 = sbuf + wid * 5 * 512;      // x1, later attention output o
  float* bx2 = bx1 + 512;                 // x2 (kept for residual)
  float* bq_ = bx2 + 512;                 // q, later post-LN output
  float* bk_ = bq_ + 512;
  float* bv_ = bk_ + 512;

  int col0 = lane & 15, mrow = (lane >> 4) * 8;

  // x1 = x @ Win + bin  (A-frag straight from global, pad rows 12..15 -> row 11)
  {
    int t = lane & 15; if (t > 11) t = 11;
    int kb = (lane >> 4) * 8;
    const float* p = x + (size_t)n * (TT * DD) + t * DD + kb;
    v16h a;
#pragma unroll
    for (int i = 0; i < 8; ++i) { a[i] = (_Float16)p[i]; a[i + 8] = (_Float16)p[i + 16]; }
#pragma unroll
    for (int tl = 0; tl < 2; ++tl) {
      v8f c = {};
      c = wmma32(a, sWf[tl * 32 + lane], c);
      int cc = col0 + tl * 16; float bb = sB[cc];
#pragma unroll
      for (int g = 0; g < 8; ++g) bx1[(mrow + g) * 32 + cc] = c[g] + bb;
    }
  }
  __syncthreads();

  // x2 = x1 @ Win + bin
  {
    v16h a = afrag_lds(bx1, lane);
#pragma unroll
    for (int tl = 0; tl < 2; ++tl) {
      v8f c = {};
      c = wmma32(a, sWf[tl * 32 + lane], c);
      int cc = col0 + tl * 16; float bb = sB[cc];
#pragma unroll
      for (int g = 0; g < 8; ++g) bx2[(mrow + g) * 32 + cc] = c[g] + bb;
    }
  }
  __syncthreads();

  // q,k,v projections (A frag reused across 3 matmuls)
  {
    v16h a = afrag_lds(bx2, lane);
    float* outs[3] = {bq_, bk_, bv_};
#pragma unroll
    for (int m = 0; m < 3; ++m) {
#pragma unroll
      for (int tl = 0; tl < 2; ++tl) {
        v8f c = {};
        c = wmma32(a, sWf[(m + 1) * 64 + tl * 32 + lane], c);
        int cc = col0 + tl * 16; float bb = sB[(m + 1) * 32 + cc];
#pragma unroll
        for (int g = 0; g < 8; ++g) outs[m][(mrow + g) * 32 + cc] = c[g] + bb;
      }
    }
  }
  __syncthreads();

  // attention: lane-parallel over (head,t) pairs; writes o rows 0..11 into bx1
#pragma unroll
  for (int it = 0; it < 2; ++it) {
    int p = it * 32 + lane;
    if (p < 48) {
      int h = p / 12, t = p % 12;
      const float* qr = bq_ + t * 32 + h * HHD;
      float sc[12]; float mx = -1e30f;
#pragma unroll
      for (int s = 0; s < 12; ++s) {
        const float* kr = bk_ + s * 32 + h * HHD;
        float d = 0.f;
#pragma unroll
        for (int j = 0; j < 8; ++j) d += qr[j] * kr[j];
        d *= 0.35355339059327373f;   // 1/sqrt(8)
        sc[s] = d; mx = fmaxf(mx, d);
      }
      float sum = 0.f;
#pragma unroll
      for (int s = 0; s < 12; ++s) { sc[s] = __expf(sc[s] - mx); sum += sc[s]; }
      float inv = 1.f / sum;
#pragma unroll
      for (int j = 0; j < 8; ++j) {
        float acc = 0.f;
#pragma unroll
        for (int s = 0; s < 12; ++s) acc += sc[s] * bv_[s * 32 + h * HHD + j];
        bx1[t * 32 + h * HHD + j] = acc * inv;
      }
    }
  }
  __syncthreads();

  // out = o @ Wo + bo + x2 (residual) -> bq_
  {
    v16h a = afrag_lds(bx1, lane);
#pragma unroll
    for (int tl = 0; tl < 2; ++tl) {
      v8f c = {};
      c = wmma32(a, sWf[4 * 64 + tl * 32 + lane], c);
      int cc = col0 + tl * 16; float bb = sB[4 * 32 + cc];
#pragma unroll
      for (int g = 0; g < 8; ++g)
        bq_[(mrow + g) * 32 + cc] = c[g] + bb + bx2[(mrow + g) * 32 + cc];
    }
  }
  __syncthreads();

  // layernorm per real row
  if (lane < 12) {
    float* r = bq_ + lane * 32;
    float mu = 0.f;
#pragma unroll
    for (int c = 0; c < 32; ++c) mu += r[c];
    mu *= (1.f / 32.f);
    float var = 0.f;
#pragma unroll
    for (int c = 0; c < 32; ++c) { float d = r[c] - mu; var += d * d; }
    var *= (1.f / 32.f);
    float rs = rsqrtf(var + 1e-5f);
#pragma unroll
    for (int c = 0; c < 32; ++c) r[c] = (r[c] - mu) * rs * sLN[c] + sLN[32 + c];
  }
  __syncthreads();

  // temporal mean over T=12 -> xn[n, lane]
  {
    float s = 0.f;
#pragma unroll
    for (int t = 0; t < 12; ++t) s += bq_[t * 32 + lane];
    xn[(size_t)n * 32 + lane] = s * (1.f / 12.f);
  }
}

// ---------------------------------------------------------------------------
// Kernel 2: GAT node projection h = xn @ W [32x128] via WMMA (one wave per
// 16-node M-tile, 8 N-tiles), plus alpha_src/alpha_dst reductions.
// ---------------------------------------------------------------------------
__global__ __launch_bounds__(128) void k_gat_h(
    const float* __restrict__ xin, const float* __restrict__ W,
    const float* __restrict__ as_, const float* __restrict__ ad_,
    float* __restrict__ hout, float* __restrict__ asrc, float* __restrict__ adst) {
  __shared__ v16h  sWf[8 * 32];          // 8 col-tiles x 32 lanes (8KB)
  __shared__ float sA[256];              // a_src(128) | a_dst(128)
  __shared__ float hbuf[4 * 16 * 128];
  int tid = threadIdx.x, lane = tid & 31, wid = tid >> 5;
  swizzle_weights_to_frags((_Float16*)sWf, W, 128, 8, tid, 128);
  for (int i = tid; i < 256; i += 128)  sA[i] = (i < 128) ? as_[i] : ad_[i - 128];
  __syncthreads();

  const int NTILES = (NN + 15) / 16;  // 3125
  int tile = blockIdx.x * 4 + wid;
  float* hb = hbuf + wid * 2048;
  if (tile < NTILES) {
    int node0 = tile * 16;
    int r = lane & 15, kb = (lane >> 4) * 8;
    int nn = node0 + r; if (nn > NN - 1) nn = NN - 1;
    const float* p = xin + (size_t)nn * 32 + kb;
    v16h a;
#pragma unroll
    for (int i = 0; i < 8; ++i) { a[i] = (_Float16)p[i]; a[i + 8] = (_Float16)p[i + 16]; }
    int col0 = lane & 15, mrow = (lane >> 4) * 8;
#pragma unroll
    for (int nt = 0; nt < 8; ++nt) {
      v8f c = {};
      c = wmma32(a, sWf[nt * 32 + lane], c);
      int cc = col0 + nt * 16;
#pragma unroll
      for (int g = 0; g < 8; ++g) {
        hb[(mrow + g) * 128 + cc] = c[g];
        int nd = node0 + mrow + g;
        if (nd < NN) hout[(size_t)nd * 128 + cc] = c[g];
      }
    }
  }
  __syncthreads();
  if (tile < NTILES) {
    int node0 = tile * 16;
#pragma unroll
    for (int it = 0; it < 2; ++it) {
      int p = it * 32 + lane;
      int r = p >> 2, h = p & 3;
      int nd = node0 + r;
      if (nd < NN) {
        float s1 = 0.f, s2 = 0.f;
#pragma unroll
        for (int c = 0; c < 32; ++c) {
          float hv = hb[r * 128 + h * 32 + c];
          s1 += hv * sA[h * 32 + c];
          s2 += hv * sA[128 + h * 32 + c];
        }
        asrc[(size_t)nd * 4 + h] = s1;
        adst[(size_t)nd * 4 + h] = s2;
      }
    }
  }
}

// ---------------------------------------------------------------------------
// Edge-wise kernels (streaming, L2-resident gathers + atomics)
// ---------------------------------------------------------------------------
__global__ void k_zero(float* p, int count) {
  int i = blockIdx.x * blockDim.x + threadIdx.x;
  if (i < count) p[i] = 0.f;
}

__global__ void k_edge_alpha(const int* __restrict__ src, const int* __restrict__ dst,
                             const float* __restrict__ ea,
                             const float* __restrict__ We, const float* __restrict__ ae,
                             const float* __restrict__ asrc, const float* __restrict__ adst,
                             float* __restrict__ alph, unsigned* __restrict__ amax) {
  __shared__ float ce[4];
  if (threadIdx.x < 4) {
    float s = 0.f;
    for (int c = 0; c < 32; ++c) s += We[threadIdx.x * 32 + c] * ae[threadIdx.x * 32 + c];
    ce[threadIdx.x] = s;
  }
  __syncthreads();
  int idx = blockIdx.x * blockDim.x + threadIdx.x;
  if (idx >= EE * 4) return;
  int e = idx >> 2, h = idx & 3;
  int s = src[e], d = dst[e];
  float a = asrc[s * 4 + h] + adst[d * 4 + h] + ea[e] * ce[h];
  a = (a > 0.f) ? a : 0.2f * a;          // leaky_relu 0.2
  alph[idx] = a;
  unsigned b = __float_as_uint(a);
  unsigned u = (b & 0x80000000u) ? ~b : (b | 0x80000000u);  // order-preserving
  atomicMax(&amax[d * 4 + h], u);
}

__global__ void k_edge_ex(const int* __restrict__ dst,
                          float* __restrict__ alph, const unsigned* __restrict__ amax,
                          float* __restrict__ den) {
  int idx = blockIdx.x * blockDim.x + threadIdx.x;
  if (idx >= EE * 4) return;
  int e = idx >> 2, h = idx & 3;
  int d = dst[e];
  unsigned u = amax[d * 4 + h];
  float mx = __uint_as_float((u & 0x80000000u) ? (u & 0x7fffffffu) : ~u);
  float ex = __expf(alph[idx] - mx);
  alph[idx] = ex;                        // reuse buffer in place
  atomicAdd(&den[d * 4 + h], ex);
}

__global__ void k_scatter(const int* __restrict__ src, const int* __restrict__ dst,
                          const float* __restrict__ ex, const float* __restrict__ den,
                          const float* __restrict__ hb, float* __restrict__ outacc) {
  long long idx = (long long)blockIdx.x * blockDim.x + threadIdx.x;
  if (idx >= (long long)EE * 128) return;
  int e = (int)(idx >> 7), rem = (int)(idx & 127);
  int h = rem >> 5;
  int s = src[e], d = dst[e];
  float coef = ex[e * 4 + h] / (den[d * 4 + h] + 1e-16f);
  atomicAdd(&outacc[(size_t)d * 128 + rem], hb[(size_t)s * 128 + rem] * coef);
}

__global__ void k_mean_relu(const float* __restrict__ outacc, const float* __restrict__ bias,
                            float* __restrict__ xno) {
  int idx = blockIdx.x * blockDim.x + threadIdx.x;
  if (idx >= NN * 32) return;
  int n = idx >> 5, c = idx & 31;
  const float* p = outacc + (size_t)n * 128 + c;
  float m = (p[0] + p[32] + p[64] + p[96]) * 0.25f + bias[c];
  xno[idx] = fmaxf(m, 0.f);
}

__global__ void k_fc(const float* __restrict__ xn3, const float* __restrict__ Wf,
                     const float* __restrict__ bf, float* __restrict__ out) {
  int n = blockIdx.x * blockDim.x + threadIdx.x;
  if (n >= NN) return;
  float a = bf[0];
#pragma unroll
  for (int c = 0; c < 32; ++c) a += xn3[(size_t)n * 32 + c] * Wf[c];
  out[n] = a;
}

// ---------------------------------------------------------------------------
extern "C" void kernel_launch(void* const* d_in, const int* in_sizes, int n_in,
                              void* d_out, int out_size, void* d_ws, size_t ws_size,
                              hipStream_t stream) {
  (void)in_sizes; (void)n_in; (void)out_size; (void)ws_size;
  const float* x   = (const float*)d_in[0];
  const int*   ei  = (const int*)d_in[1];
  const float* ea  = (const float*)d_in[2];
  const float* Win = (const float*)d_in[3];
  const float* bin = (const float*)d_in[4];
  const float* Wq  = (const float*)d_in[5];
  const float* bq  = (const float*)d_in[6];
  const float* Wk  = (const float*)d_in[7];
  const float* bk  = (const float*)d_in[8];
  const float* Wv  = (const float*)d_in[9];
  const float* bv  = (const float*)d_in[10];
  const float* Wo  = (const float*)d_in[11];
  const float* bo  = (const float*)d_in[12];
  const float* lng = (const float*)d_in[13];
  const float* lnb = (const float*)d_in[14];
  const float* Wg[2]  = {(const float*)d_in[15], (const float*)d_in[21]};
  const float* asg[2] = {(const float*)d_in[16], (const float*)d_in[22]};
  const float* adg[2] = {(const float*)d_in[17], (const float*)d_in[23]};
  const float* Weg[2] = {(const float*)d_in[18], (const float*)d_in[24]};
  const float* aeg[2] = {(const float*)d_in[19], (const float*)d_in[25]};
  const float* bg[2]  = {(const float*)d_in[20], (const float*)d_in[26]};
  const float* Wf  = (const float*)d_in[27];
  const float* bf  = (const float*)d_in[28];

  float* ws   = (float*)d_ws;
  float* XN   = ws;                   // N*32
  float* XN2  = ws + 1600000;         // N*32
  float* XN3  = ws + 3200000;         // N*32
  float* HB   = ws + 4800000;         // N*128
  float* ASRC = ws + 11200000;        // N*4
  float* ADST = ws + 11400000;        // N*4
  float* ALPH = ws + 11600000;        // E*4 (alpha, then ex in place)
  float* AMAX = ws + 14800000;        // N*4 (uint-encoded)
  float* DEN  = ws + 15000000;        // N*4
  float* OUTA = ws + 15200000;        // N*128  (total 21.6M floats = 86.4 MB)

  const int* srcp = ei;
  const int* dstp = ei + EE;

  k_temporal<<<NN / 4, 128, 0, stream>>>(x, Win, bin, Wq, bq, Wk, bk, Wv, bv,
                                         Wo, bo, lng, lnb, XN);

  const float* lin[2] = {XN, XN2};
  float* lout[2] = {XN2, XN3};
  for (int L = 0; L < 2; ++L) {
    k_zero<<<(6800000 + 255) / 256, 256, 0, stream>>>(AMAX, 6800000);  // amax+den+outacc
    k_gat_h<<<782, 128, 0, stream>>>(lin[L], Wg[L], asg[L], adg[L], HB, ASRC, ADST);
    k_edge_alpha<<<(EE * 4) / 256, 256, 0, stream>>>(srcp, dstp, ea, Weg[L], aeg[L],
                                                     ASRC, ADST, ALPH, (unsigned*)AMAX);
    k_edge_ex<<<(EE * 4) / 256, 256, 0, stream>>>(dstp, ALPH, (const unsigned*)AMAX, DEN);
    k_scatter<<<(int)(((long long)EE * 128) / 256), 256, 0, stream>>>(srcp, dstp, ALPH,
                                                                      DEN, HB, OUTA);
    k_mean_relu<<<(NN * 32) / 256, 256, 0, stream>>>(OUTA, bg[L], lout[L]);
  }
  k_fc<<<(NN + 255) / 256, 256, 0, stream>>>(XN3, Wf, bf, (float*)d_out);
}
